// Block_67611375173664
// MI455X (gfx1250) — compile-verified, matrix operand order checked by reading, not run on hardware
//
#include <hip/hip_runtime.h>
#include <hip/hip_bf16.h>
#include <math.h>

// ---------------------------------------------------------------------------
// Transformer block forward for MI455X (gfx1250, wave32, WMMA).
// B=4, N=2048, C=768, H=12, DH=64, HID=3072.  All GEMMs + attention use
// v_wmma_f32_16x16x32_bf16 with fp32 accumulation.  GEMM tiles are 64x128,
// A tiles staged to LDS via GLOBAL_LOAD_ASYNC_TO_LDS_B128 when available.
// ---------------------------------------------------------------------------

#define BT  4
#define NT  2048
#define CT  768
#define HT  12
#define DHT 64
#define HIDT 3072
#define MT  (BT * NT)   // 8192 rows

typedef __attribute__((ext_vector_type(16))) __bf16 v16bf;
typedef __attribute__((ext_vector_type(8)))  float  v8f;

union FragBF { v16bf v; uint4 q[2]; };

#if defined(__gfx1250__) && \
    __has_builtin(__builtin_amdgcn_global_load_async_to_lds_b128) && \
    __has_builtin(__builtin_amdgcn_s_wait_asynccnt)
#define USE_ASYNC_LDS 1
// builtin signature: (v4i32 AS1* src, v4i32 AS3* dst, imm i32 offset, imm i32 cpol)
typedef int v4i32 __attribute__((vector_size(16)));
typedef __attribute__((address_space(1))) v4i32* as1_v4i32p;
typedef __attribute__((address_space(3))) v4i32* as3_v4i32p;
#else
#define USE_ASYNC_LDS 0
#endif

__device__ __forceinline__ v8f wmma_bf16(v16bf a, v16bf b, v8f c) {
  // D = A(16x32 bf16) * B(32x16 bf16) + C(16x16 f32)
  return __builtin_amdgcn_wmma_f32_16x16x32_bf16(
      /*neg_a=*/false, a, /*neg_b=*/false, b,
      /*c_mod=*/(short)0, c, /*reuse_a=*/false, /*reuse_b=*/false);
}

__device__ __forceinline__ v8f zero8() {
  v8f z;
#pragma unroll
  for (int i = 0; i < 8; ++i) z[i] = 0.0f;
  return z;
}

// fp32 -> bf16 (round to nearest even), stored as raw u16
__device__ __forceinline__ unsigned short f2bf(float f) {
  union { float f; unsigned int u; } c; c.f = f;
  unsigned int u = c.u;
  unsigned int r = u + 0x7FFFu + ((u >> 16) & 1u);
  if ((u & 0x7F800000u) == 0x7F800000u) r = u;   // inf / NaN: truncate
  return (unsigned short)(r >> 16);
}

// ---------------------------------------------------------------------------
// Elementwise fp32 -> bf16 cast (weights)
// ---------------------------------------------------------------------------
__global__ __launch_bounds__(256) void cast_bf16_kernel(
    const float* __restrict__ src, unsigned short* __restrict__ dst, int n) {
  int i = blockIdx.x * 256 + threadIdx.x;
  if (i < n) dst[i] = f2bf(src[i]);
}

// ---------------------------------------------------------------------------
// LayerNorm over C=768, one block per row, output bf16
// ---------------------------------------------------------------------------
__global__ __launch_bounds__(256) void ln_bf16_kernel(
    const float* __restrict__ x, const float* __restrict__ g,
    const float* __restrict__ b, unsigned short* __restrict__ out) {
  __shared__ float2 red[256];
  const int tid = threadIdx.x;
  const size_t base = (size_t)blockIdx.x * CT;
  float v0 = x[base + tid];
  float v1 = x[base + tid + 256];
  float v2 = x[base + tid + 512];
  red[tid] = make_float2(v0 + v1 + v2, v0 * v0 + v1 * v1 + v2 * v2);
  __syncthreads();
#pragma unroll
  for (int s = 128; s > 0; s >>= 1) {
    if (tid < s) {
      red[tid].x += red[tid + s].x;
      red[tid].y += red[tid + s].y;
    }
    __syncthreads();
  }
  const float mu  = red[0].x * (1.0f / CT);
  const float var = red[0].y * (1.0f / CT) - mu * mu;
  const float inv = rsqrtf(var + 1e-5f);
  out[base + tid]       = f2bf((v0 - mu) * inv * g[tid]       + b[tid]);
  out[base + tid + 256] = f2bf((v1 - mu) * inv * g[tid + 256] + b[tid + 256]);
  out[base + tid + 512] = f2bf((v2 - mu) * inv * g[tid + 512] + b[tid + 512]);
}

// ---------------------------------------------------------------------------
// Generic 64x128-tile bf16 GEMM:  out = A[M,K] * W[K,N]  (+ epilogue)
//   MODE 0: QKV scatter -> q/k/v bf16 buffers [B,H,N,DH]
//   MODE 1: + bias + residual -> fp32 out
//   MODE 2: + bias, exact GELU -> bf16 out
// 8 waves/block; wave w owns row-block rb=w>>1 and 4 col-blocks (w&1)*4..+3.
// A tile staged via async-LDS copy when the builtin exists.
// ---------------------------------------------------------------------------
template <int MODE>
__global__ __launch_bounds__(256) void gemm64x128_kernel(
    const unsigned short* __restrict__ A, const unsigned short* __restrict__ W,
    const float* __restrict__ bias, const float* __restrict__ res,
    float* __restrict__ outF, unsigned short* __restrict__ outBf,
    unsigned short* __restrict__ qOut, unsigned short* __restrict__ kOut,
    unsigned short* __restrict__ vOut, int Ntot, int Ktot) {
  __shared__ __align__(16) unsigned short As[64][32];    // [m][k]
  __shared__ __align__(16) unsigned short Bt[128][40];   // [n][k] (transposed)

  const int tid   = threadIdx.x;
  const int lane  = tid & 31;
  const int w     = tid >> 5;
  const int rb    = w >> 1;           // 0..3 row block
  const int cbA   = (w & 1) * 4;      // col blocks cbA..cbA+3
  const int mbase = blockIdx.y * 64;
  const int nbase = blockIdx.x * 128;
  const int lmod  = lane & 15;
  const int khalf = (lane >> 4) * 8;  // wave32 WMMA half-lane K offset

  v8f acc[4];
#pragma unroll
  for (int t = 0; t < 4; ++t) acc[t] = zero8();

  // per-thread staging coordinates
  const int ar = tid >> 2, ac = (tid & 3) * 8;          // A: 64x32, 8 el/thread
  const int bk = tid >> 3, bn0 = (tid & 7) * 16;        // B: 32x128, 16 el/thread

  for (int k0 = 0; k0 < Ktot; k0 += 32) {
    __syncthreads();
    // ---- stage A tile 64x32 ----
    {
      const size_t off = (size_t)(mbase + ar) * Ktot + k0 + ac;
#if USE_ASYNC_LDS
      __builtin_amdgcn_global_load_async_to_lds_b128(
          (as1_v4i32p)(A + off), (as3_v4i32p)&As[ar][ac], 0, 0);
#else
      *(uint4*)&As[ar][ac] = *(const uint4*)(A + off);
#endif
      if (k0 + 32 < Ktot)
        __builtin_prefetch(A + off + 32, 0, 0);   // global_prefetch_b8
    }
    // ---- stage B tile 32x128, transposed into LDS -> [n][k] ----
    {
      const size_t off = (size_t)(k0 + bk) * Ntot + nbase + bn0;
#pragma unroll
      for (int h = 0; h < 2; ++h) {
        uint4 pk = *(const uint4*)(W + off + h * 8);
        unsigned int pw[4] = {pk.x, pk.y, pk.z, pk.w};
#pragma unroll
        for (int j = 0; j < 4; ++j) {
          Bt[bn0 + h * 8 + 2 * j][bk]     = (unsigned short)(pw[j] & 0xFFFFu);
          Bt[bn0 + h * 8 + 2 * j + 1][bk] = (unsigned short)(pw[j] >> 16);
        }
      }
      if (k0 + 32 < Ktot)
        __builtin_prefetch(W + off + (size_t)32 * Ntot, 0, 0);
    }
#if USE_ASYNC_LDS
    __builtin_amdgcn_s_wait_asynccnt(0);
#endif
    __syncthreads();

    FragBF a;
    a.q[0] = *(const uint4*)&As[rb * 16 + lmod][khalf];
    a.q[1] = *(const uint4*)&As[rb * 16 + lmod][khalf + 16];
#pragma unroll
    for (int t = 0; t < 4; ++t) {
      FragBF b;
      b.q[0] = *(const uint4*)&Bt[(cbA + t) * 16 + lmod][khalf];
      b.q[1] = *(const uint4*)&Bt[(cbA + t) * 16 + lmod][khalf + 16];
      acc[t] = wmma_bf16(a.v, b.v, acc[t]);
    }
  }

  // epilogue; C/D layout: VGPR r -> M=r (lanes 0-15) / M=r+8 (lanes 16-31)
  const int mloc = rb * 16 + ((lane >> 4) << 3);
#pragma unroll
  for (int r = 0; r < 8; ++r) {
    const int mg = mbase + mloc + r;
#pragma unroll
    for (int t = 0; t < 4; ++t) {
      const int ng = nbase + (cbA + t) * 16 + lmod;
      const float val = acc[t][r];
      if (MODE == 0) {
        // scatter to q/k/v: n -> (3, H, DH) ; m -> (b, token)
        const int i3 = ng / 768, rem = ng % 768;
        const int hh = rem >> 6, d = rem & 63;
        const int bb = mg >> 11, nn = mg & 2047;
        unsigned short* dst = (i3 == 0) ? qOut : ((i3 == 1) ? kOut : vOut);
        dst[(((size_t)bb * HT + hh) * NT + nn) * DHT + d] = f2bf(val);
      } else if (MODE == 1) {
        const size_t o = (size_t)mg * Ntot + ng;
        outF[o] = val + bias[ng] + res[o];
      } else {
        const float z = val + bias[ng];
        const float gl = 0.5f * z * (1.0f + erff(z * 0.70710678118654752f));
        outBf[(size_t)mg * Ntot + ng] = f2bf(gl);
      }
    }
  }
}

// ---------------------------------------------------------------------------
// Fused flash-style attention.  One block = one (b,h) x 64 query rows.
// S = Q K^T (WMMA) -> LDS, online softmax (running max/sum), O += P V (WMMA).
// q/k/v bf16 [B,H,N,DH]; output o bf16 [B,N,C].
// ---------------------------------------------------------------------------
__global__ __launch_bounds__(256) void attn_kernel(
    const unsigned short* __restrict__ q, const unsigned short* __restrict__ k,
    const unsigned short* __restrict__ v, unsigned short* __restrict__ o) {
  __shared__ __align__(16) float          Ssh[64][68];
  __shared__ __align__(16) unsigned short Psh[64][72];
  __shared__ __align__(16) unsigned short Vts[64][72];
  __shared__ float rowM[64], rowL[64], rowA[64];

  const int tid  = threadIdx.x;
  const int lane = tid & 31;
  const int w    = tid >> 5;
  const int qb   = w >> 1;            // query 16-row block 0..3
  const int cbA  = (w & 1) * 2;       // 16-col blocks cbA, cbA+1
  const int bh   = blockIdx.x >> 5;   // (b*H + h)
  const int qt   = blockIdx.x & 31;   // query tile (64 rows)
  const int lmod  = lane & 15;
  const int khalf = (lane >> 4) * 8;
  const float scale = 0.125f;         // DH^-0.5

  if (tid < 64) { rowM[tid] = -1e30f; rowL[tid] = 0.0f; }

  // Q fragments (A matrix, 16x32 per slice), loaded once.
  FragBF qa0, qa1;
  {
    const unsigned short* qr =
        q + ((size_t)bh * NT + qt * 64 + qb * 16 + lmod) * DHT;
    qa0.q[0] = *(const uint4*)(qr + khalf);
    qa0.q[1] = *(const uint4*)(qr + khalf + 16);
    qa1.q[0] = *(const uint4*)(qr + 32 + khalf);
    qa1.q[1] = *(const uint4*)(qr + 32 + khalf + 16);
  }

  v8f oac0 = zero8(), oac1 = zero8();
  const int mrow = qb * 16 + ((lane >> 4) << 3);

  for (int kt = 0; kt < NT / 64; ++kt) {
    __syncthreads();   // prior P/Vt/S consumers done

    // --- S subtiles: Q (16x64) * K^T ---
#pragma unroll
    for (int t = 0; t < 2; ++t) {
      const unsigned short* kr =
          k + ((size_t)bh * NT + kt * 64 + (cbA + t) * 16 + lmod) * DHT;
      FragBF kb0, kb1;
      kb0.q[0] = *(const uint4*)(kr + khalf);
      kb0.q[1] = *(const uint4*)(kr + khalf + 16);
      kb1.q[0] = *(const uint4*)(kr + 32 + khalf);
      kb1.q[1] = *(const uint4*)(kr + 32 + khalf + 16);
      v8f s = zero8();
      s = wmma_bf16(qa0.v, kb0.v, s);
      s = wmma_bf16(qa1.v, kb1.v, s);
#pragma unroll
      for (int r = 0; r < 8; ++r)
        Ssh[mrow + r][(cbA + t) * 16 + lmod] = s[r] * scale;
    }

    // --- stage V tile transposed: Vts[d][key] ---
    for (int it = tid; it < 512; it += 256) {
      const int key = it >> 3, dc = (it & 7) * 8;
      uint4 pk = *(const uint4*)(v + ((size_t)bh * NT + kt * 64 + key) * DHT + dc);
      unsigned int pw[4] = {pk.x, pk.y, pk.z, pk.w};
#pragma unroll
      for (int j = 0; j < 4; ++j) {
        Vts[dc + 2 * j][key]     = (unsigned short)(pw[j] & 0xFFFFu);
        Vts[dc + 2 * j + 1][key] = (unsigned short)(pw[j] >> 16);
      }
    }
    __syncthreads();

    // --- online softmax, one thread per query row ---
    if (tid < 64) {
      const float mOld = rowM[tid];
      float mx = mOld;
      for (int c = 0; c < 64; ++c) mx = fmaxf(mx, Ssh[tid][c]);
      const float a = __expf(mOld - mx);
      float sum = 0.0f;
      for (int c = 0; c < 64; ++c) {
        const float p = __expf(Ssh[tid][c] - mx);
        Psh[tid][c] = f2bf(p);
        sum += p;
      }
      rowL[tid] = rowL[tid] * a + sum;
      rowM[tid] = mx;
      rowA[tid] = a;
    }
    __syncthreads();

    // --- rescale O, then O += P * V ---
#pragma unroll
    for (int r = 0; r < 8; ++r) {
      const float a = rowA[mrow + r];
      oac0[r] *= a;
      oac1[r] *= a;
    }
    FragBF pa0, pa1;
    pa0.q[0] = *(const uint4*)&Psh[qb * 16 + lmod][khalf];
    pa0.q[1] = *(const uint4*)&Psh[qb * 16 + lmod][khalf + 16];
    pa1.q[0] = *(const uint4*)&Psh[qb * 16 + lmod][32 + khalf];
    pa1.q[1] = *(const uint4*)&Psh[qb * 16 + lmod][32 + khalf + 16];
#pragma unroll
    for (int t = 0; t < 2; ++t) {
      FragBF vb0, vb1;
      const int dr = (cbA + t) * 16 + lmod;
      vb0.q[0] = *(const uint4*)&Vts[dr][khalf];
      vb0.q[1] = *(const uint4*)&Vts[dr][khalf + 16];
      vb1.q[0] = *(const uint4*)&Vts[dr][32 + khalf];
      vb1.q[1] = *(const uint4*)&Vts[dr][32 + khalf + 16];
      v8f& oa = t ? oac1 : oac0;
      oa = wmma_bf16(pa0.v, vb0.v, oa);
      oa = wmma_bf16(pa1.v, vb1.v, oa);
    }
  }

  // --- finalize: O /= l, write o[b, token, h*64 + d] (bf16) ---
  const int bb = bh / HT, hh = bh % HT;
#pragma unroll
  for (int r = 0; r < 8; ++r) {
    const int row = mrow + r;
    const float il = 1.0f / rowL[row];
    const int tok = qt * 64 + row;
#pragma unroll
    for (int t = 0; t < 2; ++t) {
      const int d = (cbA + t) * 16 + lmod;
      const float val = (t ? oac1[r] : oac0[r]) * il;
      o[((size_t)bb * NT + tok) * CT + hh * DHT + d] = f2bf(val);
    }
  }
}

// ---------------------------------------------------------------------------
// Host launcher
// ---------------------------------------------------------------------------
extern "C" void kernel_launch(void* const* d_in, const int* in_sizes, int n_in,
                              void* d_out, int out_size, void* d_ws, size_t ws_size,
                              hipStream_t stream) {
  (void)in_sizes; (void)n_in; (void)out_size; (void)ws_size;
  const float* x      = (const float*)d_in[0];
  const float* ln1_g  = (const float*)d_in[1];
  const float* ln1_b  = (const float*)d_in[2];
  const float* w_qkv  = (const float*)d_in[3];
  const float* w_proj = (const float*)d_in[4];
  const float* b_proj = (const float*)d_in[5];
  const float* ln2_g  = (const float*)d_in[6];
  const float* ln2_b  = (const float*)d_in[7];
  const float* w1     = (const float*)d_in[8];
  const float* b1     = (const float*)d_in[9];
  const float* w2     = (const float*)d_in[10];
  const float* b2     = (const float*)d_in[11];
  float* out = (float*)d_out;

  // workspace layout (256B aligned slabs)
  char* ws = (char*)d_ws;
  size_t off = 0;
  auto alloc = [&](size_t bytes) -> char* {
    char* p = ws + off;
    off += (bytes + 255) & ~(size_t)255;
    return p;
  };
  unsigned short* wqkv_bf  = (unsigned short*)alloc((size_t)CT * 3 * CT * 2);
  unsigned short* wproj_bf = (unsigned short*)alloc((size_t)CT * CT * 2);
  unsigned short* w1_bf    = (unsigned short*)alloc((size_t)CT * HIDT * 2);
  unsigned short* w2_bf    = (unsigned short*)alloc((size_t)HIDT * CT * 2);
  unsigned short* h_bf     = (unsigned short*)alloc((size_t)MT * CT * 2);     // ln1, reused for ln2
  unsigned short* qkvo     = (unsigned short*)alloc((size_t)MT * CT * 2 * 4); // q,k,v,o; reused as m1
  float*          x1       = (float*)alloc((size_t)MT * CT * 4);

  unsigned short* q_bf = qkvo;
  unsigned short* k_bf = qkvo + (size_t)MT * CT;
  unsigned short* v_bf = qkvo + (size_t)2 * MT * CT;
  unsigned short* o_bf = qkvo + (size_t)3 * MT * CT;
  unsigned short* m1_bf = qkvo;          // aliases q/k/v/o after they are dead
  unsigned short* h2_bf = h_bf;          // aliases h after QKV GEMM

  // 1) weight casts fp32 -> bf16
  {
    const int n0 = CT * 3 * CT, n1 = CT * CT, n2 = CT * HIDT;
    cast_bf16_kernel<<<(n0 + 255) / 256, 256, 0, stream>>>(w_qkv, wqkv_bf, n0);
    cast_bf16_kernel<<<(n1 + 255) / 256, 256, 0, stream>>>(w_proj, wproj_bf, n1);
    cast_bf16_kernel<<<(n2 + 255) / 256, 256, 0, stream>>>(w1, w1_bf, n2);
    cast_bf16_kernel<<<(n2 + 255) / 256, 256, 0, stream>>>(w2, w2_bf, n2);
  }
  // 2) LN1 -> h (bf16)
  ln_bf16_kernel<<<MT, 256, 0, stream>>>(x, ln1_g, ln1_b, h_bf);
  // 3) QKV GEMM with scatter epilogue
  gemm64x128_kernel<0><<<dim3(3 * CT / 128, MT / 64), 256, 0, stream>>>(
      h_bf, wqkv_bf, nullptr, nullptr, nullptr, nullptr, q_bf, k_bf, v_bf,
      3 * CT, CT);
  // 4) fused attention -> o (bf16)
  attn_kernel<<<BT * HT * (NT / 64), 256, 0, stream>>>(q_bf, k_bf, v_bf, o_bf);
  // 5) proj GEMM + bias + residual(x) -> x1 (fp32)
  gemm64x128_kernel<1><<<dim3(CT / 128, MT / 64), 256, 0, stream>>>(
      o_bf, wproj_bf, b_proj, x, x1, nullptr, nullptr, nullptr, nullptr,
      CT, CT);
  // 6) LN2 -> h2 (bf16)
  ln_bf16_kernel<<<MT, 256, 0, stream>>>(x1, ln2_g, ln2_b, h2_bf);
  // 7) MLP1 GEMM + bias + exact GELU -> m1 (bf16)
  gemm64x128_kernel<2><<<dim3(HIDT / 128, MT / 64), 256, 0, stream>>>(
      h2_bf, w1_bf, b1, nullptr, nullptr, m1_bf, nullptr, nullptr, nullptr,
      HIDT, CT);
  // 8) MLP2 GEMM + bias + residual(x1) -> out (fp32)
  gemm64x128_kernel<1><<<dim3(CT / 128, MT / 64), 256, 0, stream>>>(
      m1_bf, w2_bf, b2, x1, out, nullptr, nullptr, nullptr, nullptr,
      CT, HIDT);
}
